// SchNet_36301063586427
// MI455X (gfx1250) — matrix operand order repaired and checked by reading, not compiled, not run on hardware
//
#include <hip/hip_runtime.h>
#include <math.h>

#define N_NODES   50000
#define N_EDGES   1200000
#define N_GRAPHS  512
#define D1        64
#define NF        92
#define NG        50
#define GC_COUNT  3
#define BN_EPS    1e-5f
#define LOG2F_    0.6931471805599453f
#define PI_OVER_CUT 0.39269908169872414f   /* pi / 8.0 */

// Base workspace (floats): out, hbuf, agg, stats(256), psum, pcnt
#define WS_BASE_FLOATS (3 * N_NODES * 64 + 256 + N_GRAPHS * 64 + N_GRAPHS)

typedef __attribute__((ext_vector_type(16))) _Float16 v16h;
typedef __attribute__((ext_vector_type(8)))  float    v8f;

// ---- async global->LDS staging (CDNA5 GLOBAL_LOAD_ASYNC_TO_LDS_B128) ------
#define USE_ASYNC_LDS_COPY 1
#if USE_ASYNC_LDS_COPY && defined(__has_builtin)
#if __has_builtin(__builtin_amdgcn_global_load_async_to_lds_b128) && \
    __has_builtin(__builtin_amdgcn_s_wait_asynccnt)
#define HAVE_ASYNC_LDS 1
#endif
#endif
#ifndef HAVE_ASYNC_LDS
#define HAVE_ASYNC_LDS 0
#endif

#if HAVE_ASYNC_LDS
typedef int v4i_ __attribute__((vector_size(16)));
typedef __attribute__((address_space(1))) v4i_ GA4;  // global (device) v4i
typedef __attribute__((address_space(3))) v4i_ LA4;  // LDS v4i
__device__ __forceinline__ void ld_lds16(const void* g, void* l) {
  __builtin_amdgcn_global_load_async_to_lds_b128((GA4*)g, (LA4*)l, 0, 0);
}
__device__ __forceinline__ void lds_copy_fence() {
  __builtin_amdgcn_s_wait_asynccnt(0);
}
#else
__device__ __forceinline__ void ld_lds16(const void* g, void* l) {
  *(uint4*)l = *(const uint4*)g;
}
__device__ __forceinline__ void lds_copy_fence() {}
#endif

__device__ __forceinline__ v8f wmma16x16x32(v16h a, v16h b, v8f c) {
  return __builtin_amdgcn_wmma_f32_16x16x32_f16(false, a, false, b, (short)0, c,
                                                false, false);
}

__device__ __forceinline__ float ssp_f(float v) {
  // softplus(v) - log2, hardware transcendentals
  return fmaxf(v, 0.0f) + __logf(1.0f + __expf(-fabsf(v))) - LOG2F_;
}

// A fragment (16x32 f16, MxK): lanes 0-15 M=0..15; VGPR0-3 K=8*half+{0..7},
// VGPR4-7 K=16+8*half+{0..7}.
__device__ __forceinline__ v16h load_A_frag(const _Float16* __restrict__ As,
                                            int lda, int k0) {
  const int lane = threadIdx.x & 31;
  const int half = lane >> 4;
  const int m    = lane & 15;
  const _Float16* p = As + m * lda + k0 + 8 * half;
  v16h a;
#pragma unroll
  for (int v = 0; v < 4; ++v) { a[2*v] = p[2*v]; a[2*v+1] = p[2*v+1]; }
  p += 16;
#pragma unroll
  for (int v = 0; v < 4; ++v) { a[8+2*v] = p[2*v]; a[9+2*v] = p[2*v+1]; }
  return a;
}

// B fragment (32x16 f16, KxN), B[k][n]=Wt[n][k], Wt row-major [n][k].
__device__ __forceinline__ v16h load_B_frag(const _Float16* __restrict__ Ws,
                                            int ldb, int k0) {
  const int lane = threadIdx.x & 31;
  const int half = lane >> 4;
  const int n    = lane & 15;
  const _Float16* p = Ws + n * ldb + k0 + 8 * half;
  v16h b;
#pragma unroll
  for (int v = 0; v < 4; ++v) { b[2*v] = p[2*v]; b[2*v+1] = p[2*v+1]; }
  p += 16;
#pragma unroll
  for (int v = 0; v < 4; ++v) { b[8+2*v] = p[2*v]; b[9+2*v] = p[2*v+1]; }
  return b;
}

// ---------------------------------------------------------------------------
// Generic fp32 -> padded f16 converter: dst[r, 0..kdst) = src[r, 0..ksrc), 0-pad
// ---------------------------------------------------------------------------
__global__ __launch_bounds__(256) void k_cvt_pad(const float* __restrict__ src,
                                                 _Float16* __restrict__ dst,
                                                 int rows, int ksrc, int kdst) {
  int t = blockIdx.x * 256 + threadIdx.x;
  if (t < rows * kdst) {
    int r = t / kdst, k = t - r * kdst;
    dst[t] = (_Float16)((k < ksrc) ? src[r * ksrc + k] : 0.0f);
  }
}

// ---------------------------------------------------------------------------
// out = relu(x @ pre_W^T + pre_b)   x:[N,92]  W16:[64,96] padded f16
// ---------------------------------------------------------------------------
__global__ __launch_bounds__(256) void k_pre_fc(const float* __restrict__ x,
                                                const _Float16* __restrict__ W16,
                                                const float* __restrict__ bias,
                                                float* __restrict__ out) {
  __shared__ _Float16 sA[128 * 96];
  __shared__ _Float16 sB[64 * 96];
  const int rbase = blockIdx.x * 128;
  for (int i = threadIdx.x; i < 128 * 96; i += 256) {
    int r = i / 96, k = i - r * 96;
    int gr = rbase + r;
    sA[i] = (_Float16)((gr < N_NODES && k < NF) ? x[gr * NF + k] : 0.0f);
  }
  for (int i = threadIdx.x; i < (64 * 96) / 8; i += 256)
    ld_lds16((const uint4*)W16 + i, (uint4*)sB + i);
  lds_copy_fence();
  __syncthreads();
  const int wave = threadIdx.x >> 5;
  const int lane = threadIdx.x & 31;
  const int half = lane >> 4;
  const int n    = lane & 15;
  const _Float16* Arow = sA + (wave * 16) * 96;
  v8f acc[4] = {};
#pragma unroll
  for (int ks = 0; ks < 3; ++ks) {
    v16h a = load_A_frag(Arow, 96, ks * 32);
#pragma unroll
    for (int nt = 0; nt < 4; ++nt) {
      v16h bf = load_B_frag(sB + nt * 16 * 96, 96, ks * 32);
      acc[nt] = wmma16x16x32(a, bf, acc[nt]);
    }
  }
#pragma unroll
  for (int r = 0; r < 8; ++r) {
    int gr = rbase + wave * 16 + r + 8 * half;
    if (gr < N_NODES) {
#pragma unroll
      for (int nt = 0; nt < 4; ++nt) {
        int col = nt * 16 + n;
        out[gr * 64 + col] = fmaxf(acc[nt][r] + bias[col], 0.0f);
      }
    }
  }
}

// ---------------------------------------------------------------------------
// out = A @ W^T   (A:[N,64] fp32, W16:[64,64] f16) -- CFConv lin1
// ---------------------------------------------------------------------------
__global__ __launch_bounds__(256) void k_node_gemm64(const float* __restrict__ A,
                                                     const _Float16* __restrict__ W16,
                                                     float* __restrict__ out) {
  __shared__ _Float16 sA[128 * 64];
  __shared__ _Float16 sB[64 * 64];
  const int rbase = blockIdx.x * 128;
  for (int i = threadIdx.x; i < 128 * 64; i += 256) {
    int r = i >> 6, k = i & 63;
    int gr = rbase + r;
    sA[i] = (_Float16)((gr < N_NODES) ? A[gr * 64 + k] : 0.0f);
  }
  for (int i = threadIdx.x; i < 512; i += 256)
    ld_lds16((const uint4*)W16 + i, (uint4*)sB + i);
  lds_copy_fence();
  __syncthreads();
  const int wave = threadIdx.x >> 5;
  const int lane = threadIdx.x & 31;
  const int half = lane >> 4;
  const int n    = lane & 15;
  const _Float16* Arow = sA + wave * 16 * 64;
  v8f acc[4] = {};
#pragma unroll
  for (int ks = 0; ks < 2; ++ks) {
    v16h a = load_A_frag(Arow, 64, ks * 32);
#pragma unroll
    for (int nt = 0; nt < 4; ++nt) {
      v16h bf = load_B_frag(sB + nt * 16 * 64, 64, ks * 32);
      acc[nt] = wmma16x16x32(a, bf, acc[nt]);
    }
  }
#pragma unroll
  for (int r = 0; r < 8; ++r) {
    int gr = rbase + wave * 16 + r + 8 * half;
    if (gr < N_NODES) {
#pragma unroll
      for (int nt = 0; nt < 4; ++nt) out[gr * 64 + nt * 16 + n] = acc[nt][r];
    }
  }
}

// ---------------------------------------------------------------------------
// Fused edge pipeline. CACHED: edge_attr pre-converted to f16 [E,64]
// (L2-resident, 153.6MB vs 192MB L2); else convert fp32 in-kernel.
// ---------------------------------------------------------------------------
template <bool CACHED>
__global__ __launch_bounds__(256) void k_edge(const float* __restrict__ attr32,
                                              const _Float16* __restrict__ attr16,
                                              const float* __restrict__ edge_weight,
                                              const int*   __restrict__ src,
                                              const int*   __restrict__ dst,
                                              const _Float16* __restrict__ W1h,
                                              const float* __restrict__ b1,
                                              const _Float16* __restrict__ W2h,
                                              const float* __restrict__ b2,
                                              const float* __restrict__ hbuf,
                                              float* __restrict__ agg) {
  __shared__ _Float16 sA [128 * 64];
  __shared__ _Float16 sT [128 * 64];
  __shared__ _Float16 sW1[64 * 64];
  __shared__ _Float16 sW2[64 * 64];
  __shared__ float sC[128];
  __shared__ int   sS[128];
  __shared__ int   sD[128];
  const int ebase = blockIdx.x * 128;
  if (CACHED) {
    const uint4* asrc = (const uint4*)(attr16 + (long)ebase * 64);
    for (int i = threadIdx.x; i < 1024; i += 256)
      ld_lds16(asrc + i, (uint4*)sA + i);
  } else {
    for (int i = threadIdx.x; i < 128 * 64; i += 256) {
      int r = i >> 6, k = i & 63;
      int e = ebase + r;
      sA[i] = (_Float16)((e < N_EDGES && k < NG) ? attr32[e * NG + k] : 0.0f);
    }
  }
  for (int i = threadIdx.x; i < 512; i += 256) {
    ld_lds16((const uint4*)W1h + i, (uint4*)sW1 + i);
    ld_lds16((const uint4*)W2h + i, (uint4*)sW2 + i);
  }
  if (threadIdx.x < 128) {
    int e = ebase + (int)threadIdx.x;
    if (e < N_EDGES) {
      sC[threadIdx.x] = 0.5f * (__cosf(edge_weight[e] * PI_OVER_CUT) + 1.0f);
      sS[threadIdx.x] = src[e];
      sD[threadIdx.x] = dst[e];
    } else {
      sC[threadIdx.x] = 0.0f; sS[threadIdx.x] = 0; sD[threadIdx.x] = -1;
    }
  }
  lds_copy_fence();
  __syncthreads();
  const int wave = threadIdx.x >> 5;
  const int lane = threadIdx.x & 31;
  const int half = lane >> 4;
  const int n    = lane & 15;
  const _Float16* Arow = sA + wave * 16 * 64;
  _Float16*       Trow = sT + wave * 16 * 64;

  v8f acc[4] = {};
#pragma unroll
  for (int ks = 0; ks < 2; ++ks) {
    v16h a = load_A_frag(Arow, 64, ks * 32);
#pragma unroll
    for (int nt = 0; nt < 4; ++nt) {
      v16h bf = load_B_frag(sW1 + nt * 16 * 64, 64, ks * 32);
      acc[nt] = wmma16x16x32(a, bf, acc[nt]);
    }
  }
#pragma unroll
  for (int r = 0; r < 8; ++r) {
    int row = r + 8 * half;
#pragma unroll
    for (int nt = 0; nt < 4; ++nt) {
      int col = nt * 16 + n;
      Trow[row * 64 + col] = (_Float16)ssp_f(acc[nt][r] + b1[col]);
    }
  }
  // GEMM2 reads only this wave's rows of sT (same-wave DS ordering).
  v8f acc2[4] = {};
#pragma unroll
  for (int ks = 0; ks < 2; ++ks) {
    v16h a = load_A_frag(Trow, 64, ks * 32);
#pragma unroll
    for (int nt = 0; nt < 4; ++nt) {
      v16h bf = load_B_frag(sW2 + nt * 16 * 64, 64, ks * 32);
      acc2[nt] = wmma16x16x32(a, bf, acc2[nt]);
    }
  }
#pragma unroll
  for (int r = 0; r < 8; ++r) {
    int row = wave * 16 + r + 8 * half;
    float cv = sC[row];
    int   sn = sS[row];
    int   dn = sD[row];
    if (dn >= 0) {
#pragma unroll
      for (int nt = 0; nt < 4; ++nt) {
        int col = nt * 16 + n;
        float wv  = (acc2[nt][r] + b2[col]) * cv;
        atomicAdd(&agg[dn * 64 + col], wv * hbuf[sn * 64 + col]);
      }
    }
  }
}

// ---------------------------------------------------------------------------
// Interaction tail: out += ssp(agg @ cfW2^T + b2) @ intW^T + bi ; BN partials
// ---------------------------------------------------------------------------
__global__ __launch_bounds__(256) void k_tail(const float* __restrict__ agg,
                                              const _Float16* __restrict__ W2h,
                                              const float* __restrict__ b2,
                                              const _Float16* __restrict__ Wih,
                                              const float* __restrict__ bi,
                                              float* __restrict__ out,
                                              float* __restrict__ csum,
                                              float* __restrict__ csumsq) {
  __shared__ _Float16 sA [128 * 64];
  __shared__ _Float16 sT [128 * 64];
  __shared__ _Float16 sW2[64 * 64];
  __shared__ _Float16 sWi[64 * 64];
  const int rbase = blockIdx.x * 128;
  for (int i = threadIdx.x; i < 128 * 64; i += 256) {
    int r = i >> 6, k = i & 63;
    int gr = rbase + r;
    sA[i] = (_Float16)((gr < N_NODES) ? agg[gr * 64 + k] : 0.0f);
  }
  for (int i = threadIdx.x; i < 512; i += 256) {
    ld_lds16((const uint4*)W2h + i, (uint4*)sW2 + i);
    ld_lds16((const uint4*)Wih + i, (uint4*)sWi + i);
  }
  lds_copy_fence();
  __syncthreads();
  const int wave = threadIdx.x >> 5;
  const int lane = threadIdx.x & 31;
  const int half = lane >> 4;
  const int n    = lane & 15;
  const _Float16* Arow = sA + wave * 16 * 64;
  _Float16*       Trow = sT + wave * 16 * 64;

  v8f acc[4] = {};
#pragma unroll
  for (int ks = 0; ks < 2; ++ks) {
    v16h a = load_A_frag(Arow, 64, ks * 32);
#pragma unroll
    for (int nt = 0; nt < 4; ++nt) {
      v16h bf = load_B_frag(sW2 + nt * 16 * 64, 64, ks * 32);
      acc[nt] = wmma16x16x32(a, bf, acc[nt]);
    }
  }
#pragma unroll
  for (int r = 0; r < 8; ++r) {
    int row = r + 8 * half;
#pragma unroll
    for (int nt = 0; nt < 4; ++nt) {
      int col = nt * 16 + n;
      Trow[row * 64 + col] = (_Float16)ssp_f(acc[nt][r] + b2[col]);
    }
  }
  v8f acc2[4] = {};
#pragma unroll
  for (int ks = 0; ks < 2; ++ks) {
    v16h a = load_A_frag(Trow, 64, ks * 32);
#pragma unroll
    for (int nt = 0; nt < 4; ++nt) {
      v16h bf = load_B_frag(sWi + nt * 16 * 64, 64, ks * 32);
      acc2[nt] = wmma16x16x32(a, bf, acc2[nt]);
    }
  }
  float ps[4] = {0.f, 0.f, 0.f, 0.f};
  float pq[4] = {0.f, 0.f, 0.f, 0.f};
#pragma unroll
  for (int r = 0; r < 8; ++r) {
    int gr = rbase + wave * 16 + r + 8 * half;
    bool valid = gr < N_NODES;
#pragma unroll
    for (int nt = 0; nt < 4; ++nt) {
      int col = nt * 16 + n;
      float v = 0.0f;
      if (valid) {
        v = out[gr * 64 + col] + acc2[nt][r] + bi[col];
        out[gr * 64 + col] = v;
      }
      ps[nt] += v;
      pq[nt] += v * v;
    }
  }
#pragma unroll
  for (int nt = 0; nt < 4; ++nt) {
    int col = nt * 16 + n;
    atomicAdd(&csum[col],   ps[nt]);
    atomicAdd(&csumsq[col], pq[nt]);
  }
}

__global__ void k_bn_fin(const float* __restrict__ sum, const float* __restrict__ sumsq,
                         const float* __restrict__ g, const float* __restrict__ b,
                         float* __restrict__ scale, float* __restrict__ shift) {
  int c = threadIdx.x;
  if (c < 64) {
    float mu  = sum[c] * (1.0f / (float)N_NODES);
    float var = sumsq[c] * (1.0f / (float)N_NODES) - mu * mu;
    float sc  = g[c] * rsqrtf(var + BN_EPS);
    scale[c] = sc;
    shift[c] = b[c] - mu * sc;
  }
}

// BN affine apply; on the last layer also fuses the global_mean_pool scatter.
__global__ __launch_bounds__(256) void k_bn_apply_pool(float* __restrict__ out,
                                                       const float* __restrict__ scale,
                                                       const float* __restrict__ shift,
                                                       int doPool,
                                                       const int* __restrict__ batch,
                                                       float* __restrict__ psum,
                                                       float* __restrict__ pcnt) {
  int t = blockIdx.x * 256 + threadIdx.x;
  if (t < N_NODES * 64) {
    int c = t & 63;
    float v = out[t] * scale[c] + shift[c];
    out[t] = v;
    if (doPool) {
      int node = t >> 6;
      int g = batch[node];
      atomicAdd(&psum[g * 64 + c], v);
      if (c == 0) atomicAdd(&pcnt[g], 1.0f);
    }
  }
}

__global__ __launch_bounds__(256) void k_zero(float* __restrict__ p, int n) {
  int t = blockIdx.x * 256 + threadIdx.x;
  if (t < n) p[t] = 0.0f;
}

// pooled = psum/max(cnt,1); h = relu(pooled @ postW^T + postb); y = h @ outW^T + outb
__global__ __launch_bounds__(256) void k_head(const float* __restrict__ psum,
                                              const float* __restrict__ pcnt,
                                              const _Float16* __restrict__ postWh,
                                              const float* __restrict__ postb,
                                              const float* __restrict__ outW,
                                              const float* __restrict__ outbp,
                                              float* __restrict__ y) {
  __shared__ _Float16 sA[128 * 64];
  __shared__ _Float16 sB[64 * 64];
  __shared__ float    sH[128 * 64];
  const int rbase = blockIdx.x * 128;
  for (int i = threadIdx.x; i < 128 * 64; i += 256) {
    int r = i >> 6, k = i & 63;
    int g = rbase + r;
    float cnt = fmaxf(pcnt[g], 1.0f);
    sA[i] = (_Float16)(psum[g * 64 + k] / cnt);
  }
  for (int i = threadIdx.x; i < 512; i += 256)
    ld_lds16((const uint4*)postWh + i, (uint4*)sB + i);
  lds_copy_fence();
  __syncthreads();
  const int wave = threadIdx.x >> 5;
  const int lane = threadIdx.x & 31;
  const int half = lane >> 4;
  const int n    = lane & 15;
  const _Float16* Arow = sA + wave * 16 * 64;
  v8f acc[4] = {};
#pragma unroll
  for (int ks = 0; ks < 2; ++ks) {
    v16h a = load_A_frag(Arow, 64, ks * 32);
#pragma unroll
    for (int nt = 0; nt < 4; ++nt) {
      v16h bf = load_B_frag(sB + nt * 16 * 64, 64, ks * 32);
      acc[nt] = wmma16x16x32(a, bf, acc[nt]);
    }
  }
#pragma unroll
  for (int r = 0; r < 8; ++r) {
    int row = wave * 16 + r + 8 * half;
#pragma unroll
    for (int nt = 0; nt < 4; ++nt) {
      int col = nt * 16 + n;
      sH[row * 64 + col] = fmaxf(acc[nt][r] + postb[col], 0.0f);
    }
  }
  __syncthreads();
  if (threadIdx.x < 128) {
    float s = outbp[0];
    const float* h = sH + threadIdx.x * 64;
#pragma unroll
    for (int c = 0; c < 64; ++c) s += h[c] * outW[c];
    y[rbase + (int)threadIdx.x] = s;
  }
}

// ---------------------------------------------------------------------------
extern "C" void kernel_launch(void* const* d_in, const int* in_sizes, int n_in,
                              void* d_out, int out_size, void* d_ws, size_t ws_size,
                              hipStream_t stream) {
  const float* x      = (const float*)d_in[0];
  const float* ew     = (const float*)d_in[1];
  const float* eattr  = (const float*)d_in[2];
  const int*   eidx   = (const int*)  d_in[3];
  const int*   batch  = (const int*)  d_in[4];
  const float* pre_W  = (const float*)d_in[5];
  const float* pre_b  = (const float*)d_in[6];
  const float* mlp_W1 = (const float*)d_in[7];
  const float* mlp_b1 = (const float*)d_in[8];
  const float* mlp_W2 = (const float*)d_in[9];
  const float* mlp_b2 = (const float*)d_in[10];
  const float* cf_W1  = (const float*)d_in[11];
  const float* cf_W2  = (const float*)d_in[12];
  const float* cf_b2  = (const float*)d_in[13];
  const float* int_W  = (const float*)d_in[14];
  const float* int_b  = (const float*)d_in[15];
  const float* bn_g   = (const float*)d_in[16];
  const float* bn_b   = (const float*)d_in[17];
  const float* post_W = (const float*)d_in[18];
  const float* post_b = (const float*)d_in[19];
  const float* out_W  = (const float*)d_in[20];
  const float* out_b  = (const float*)d_in[21];
  float* y = (float*)d_out;

  // ---- workspace layout ----
  float* ws    = (float*)d_ws;
  float* out   = ws;                       // N*64
  float* hbuf  = out  + N_NODES * 64;      // N*64
  float* agg   = hbuf + N_NODES * 64;      // N*64
  float* csum  = agg  + N_NODES * 64;      // 64
  float* csq   = csum + 64;                // 64
  float* scale = csq  + 64;                // 64
  float* shift = scale + 64;               // 64
  float* psum  = shift + 64;               // G*64
  float* pcnt  = psum + N_GRAPHS * 64;     // G

  // f16 weights (padded): after base floats
  _Float16* preW16 = (_Float16*)(ws + WS_BASE_FLOATS);  // 64*96
  _Float16* mW1h   = preW16 + 64 * 96;                  // 3 * 4096
  _Float16* mW2h   = mW1h + 3 * 4096;
  _Float16* cW1h   = mW2h + 3 * 4096;
  _Float16* cW2h   = cW1h + 3 * 4096;
  _Float16* iWh    = cW2h + 3 * 4096;
  _Float16* pWh    = iWh  + 3 * 4096;                   // 4096
  _Float16* attr16 = pWh  + 4096;                       // optional E*64

  const size_t need_base   = (size_t)WS_BASE_FLOATS * 4 + 71680 * 2;
  const size_t need_cached = need_base + (size_t)N_EDGES * 64 * 2;
  const bool cached = ws_size >= need_cached;   // deterministic per-launch

  const int* src = eidx;
  const int* dstp = eidx + N_EDGES;

  dim3 blk(256);
  const int nodeBlocks = (N_NODES + 127) / 128;   // 391
  const int edgeBlocks = (N_EDGES + 127) / 128;   // 9375
  const int ewBlocks   = (N_NODES * 64 + 255) / 256;

  // ---- weight (and optionally edge_attr) fp32 -> f16 conversion ----
  k_cvt_pad<<<(64 * 96 + 255) / 256, blk, 0, stream>>>(pre_W, preW16, 64, NF, 96);
  for (int l = 0; l < GC_COUNT; ++l) {
    k_cvt_pad<<<(4096 + 255) / 256, blk, 0, stream>>>(mlp_W1 + l * 64 * NG, mW1h + l * 4096, 64, NG, 64);
    k_cvt_pad<<<(4096 + 255) / 256, blk, 0, stream>>>(mlp_W2 + l * 4096,    mW2h + l * 4096, 64, 64, 64);
    k_cvt_pad<<<(4096 + 255) / 256, blk, 0, stream>>>(cf_W1 + l * 4096,     cW1h + l * 4096, 64, 64, 64);
    k_cvt_pad<<<(4096 + 255) / 256, blk, 0, stream>>>(cf_W2 + l * 4096,     cW2h + l * 4096, 64, 64, 64);
    k_cvt_pad<<<(4096 + 255) / 256, blk, 0, stream>>>(int_W + l * 4096,     iWh  + l * 4096, 64, 64, 64);
  }
  k_cvt_pad<<<(4096 + 255) / 256, blk, 0, stream>>>(post_W, pWh, 64, 64, 64);
  if (cached) {
    // edge_attr cached once as f16 [E,64]; 153.6MB ~ L2-resident on reuse
    k_cvt_pad<<<(N_EDGES * 64 + 255) / 256, blk, 0, stream>>>(eattr, attr16,
                                                              N_EDGES, NG, 64);
  }

  k_zero<<<(N_GRAPHS * 64 + N_GRAPHS + 255) / 256, blk, 0, stream>>>(
      psum, N_GRAPHS * 64 + N_GRAPHS);  // psum + pcnt contiguous

  k_pre_fc<<<nodeBlocks, blk, 0, stream>>>(x, preW16, pre_b, out);

  for (int l = 0; l < GC_COUNT; ++l) {
    k_zero<<<ewBlocks, blk, 0, stream>>>(agg, N_NODES * 64);
    k_zero<<<1, blk, 0, stream>>>(csum, 128);  // csum + csq contiguous
    k_node_gemm64<<<nodeBlocks, blk, 0, stream>>>(out, cW1h + l * 4096, hbuf);
    if (cached) {
      k_edge<true><<<edgeBlocks, blk, 0, stream>>>(eattr, attr16, ew, src, dstp,
          mW1h + l * 4096, mlp_b1 + l * 64, mW2h + l * 4096, mlp_b2 + l * 64,
          hbuf, agg);
    } else {
      k_edge<false><<<edgeBlocks, blk, 0, stream>>>(eattr, attr16, ew, src, dstp,
          mW1h + l * 4096, mlp_b1 + l * 64, mW2h + l * 4096, mlp_b2 + l * 64,
          hbuf, agg);
    }
    k_tail<<<nodeBlocks, blk, 0, stream>>>(agg, cW2h + l * 4096, cf_b2 + l * 64,
        iWh + l * 4096, int_b + l * 64, out, csum, csq);
    k_bn_fin<<<1, 64, 0, stream>>>(csum, csq, bn_g + l * 64, bn_b + l * 64,
                                   scale, shift);
    k_bn_apply_pool<<<ewBlocks, blk, 0, stream>>>(out, scale, shift,
        (l == GC_COUNT - 1) ? 1 : 0, batch, psum, pcnt);
  }

  k_head<<<N_GRAPHS / 128, blk, 0, stream>>>(psum, pcnt, pWh, post_b,
                                             out_W, out_b, y);
}
